// LSTMModel_46806553592164
// MI455X (gfx1250) — matrix operand order repaired and checked by reading, not compiled, not run on hardware
//
#include <hip/hip_runtime.h>
#include <hip/hip_bf16.h>

// ---------------- types ----------------
typedef float  v2f  __attribute__((ext_vector_type(2)));
typedef float  v8f  __attribute__((ext_vector_type(8)));
typedef __bf16 v16bf __attribute__((ext_vector_type(16)));

#define BB 32
#define TT 512
#define WDIM 300
#define HH 256
#define MM (BB*TT)          // 16384 rows
#define G4H 1024            // 4*H

__device__ __forceinline__ v8f wmma_f32_k4(v2f a, v2f b, v8f c) {
  // D = A(16x4,f32) * B(4x16,f32) + C(16x16,f32)
  return __builtin_amdgcn_wmma_f32_16x16x4_f32(false, a, false, b, (short)0, c, false, false);
}
__device__ __forceinline__ v8f wmma_bf16_k32(v16bf a, v16bf b, v8f c) {
  // D = A(16x32,bf16) * B(32x16,bf16) + C(16x16,f32)
  return __builtin_amdgcn_wmma_f32_16x16x32_bf16(false, a, false, b, (short)0, c, false, false);
}

__device__ __forceinline__ int perm_col(int n) {
  // n = 64*w + 16*s + l  ->  original column s*256 + 16*w + l
  return ((n >> 4) & 3) * HH + (n >> 6) * 16 + (n & 15);
}
__device__ __forceinline__ float sigmoidf_(float x) { return 1.0f / (1.0f + expf(-x)); }

// ---------------- kernel 0: weight permute / convert ----------------
__global__ void k_prep(const float* __restrict__ Wih_f, const float* __restrict__ Whh_f,
                       const float* __restrict__ bl_f,
                       const float* __restrict__ Wih_b, const float* __restrict__ Whh_b,
                       const float* __restrict__ bl_b,
                       float* __restrict__ WihP,     // [2][256][1024] f32, gate-interleaved cols
                       __bf16* __restrict__ WhhPT,   // [2][1024][256] bf16, transposed + permuted
                       float* __restrict__ blP) {    // [2][1024]
  const int stride = blockDim.x * gridDim.x;
  const int tid0 = blockIdx.x * blockDim.x + threadIdx.x;
  for (int i = tid0; i < 2 * HH * G4H; i += stride) {
    int dir = i >> 18, r = i & ((1 << 18) - 1);
    int k = r >> 10, n = r & (G4H - 1);
    const float* src = dir ? Wih_b : Wih_f;
    WihP[i] = src[k * G4H + perm_col(n)];
  }
  for (int i = tid0; i < 2 * G4H * HH; i += stride) {
    int dir = i >> 18, r = i & ((1 << 18) - 1);
    int n = r >> 8, k = r & (HH - 1);
    const float* src = dir ? Whh_b : Whh_f;
    WhhPT[i] = (__bf16)src[k * G4H + perm_col(n)];
  }
  for (int i = tid0; i < 2 * G4H; i += stride) {
    int dir = i >> 10, n = i & (G4H - 1);
    blP[i] = (dir ? bl_b : bl_f)[perm_col(n)];
  }
}

// ---------------- kernel 1: z = relu(emb[x] @ W_in + b_in) ----------------
// 128 threads = 4 waves; the 16x300 gathered A-tile is staged once in LDS and
// shared by all 4 waves (each wave owns a distinct 64-column group of N=256).
__global__ __launch_bounds__(128) void k_embed_gemm(
    const float* __restrict__ emb, const float* __restrict__ Win,
    const float* __restrict__ bin, const int* __restrict__ x,
    float* __restrict__ z) {
  __shared__ float et[16][308];  // pitch 308: 16B-aligned rows, conflict-free frags
  const int tid = threadIdx.x;
  const int w = tid >> 5, lane = tid & 31, l16 = lane & 15, hi = lane >> 4;
  const int m0 = blockIdx.x * 16;
  const int n0 = w * 64;
  for (int r = 0; r < 16; ++r) {
    const long arow = (long)x[m0 + r] * WDIM;
    for (int cc = tid; cc < WDIM; cc += 128) et[r][cc] = emb[arow + cc];
  }
  __syncthreads();
  v8f acc[4] = {};
  for (int k0 = 0; k0 < WDIM; k0 += 4) {
    const int ka = k0 + 2 * hi;
    v2f a = *(const v2f*)&et[l16][ka];
#pragma unroll
    for (int s = 0; s < 4; ++s) {
      const int col = n0 + 16 * s + l16;
      v2f b;
      b.x = Win[(size_t)ka * HH + col];
      b.y = Win[(size_t)(ka + 1) * HH + col];
      acc[s] = wmma_f32_k4(a, b, acc[s]);
    }
  }
#pragma unroll
  for (int s = 0; s < 4; ++s) {
    const int col = n0 + 16 * s + l16;
    const float bias = bin[col];
#pragma unroll
    for (int vi = 0; vi < 8; ++vi) {
      const int row = m0 + vi + 8 * hi;
      float v = acc[s][vi] + bias;
      z[(size_t)row * HH + col] = v > 0.f ? v : 0.f;
    }
  }
}

// ---------------- kernel 2: G[dir] = z @ WihP[dir] + blP[dir] ----------------
// 128 threads = 4 waves sharing one LDS-staged 16x256 A-tile of z.
__global__ __launch_bounds__(128) void k_gate_gemm(
    const float* __restrict__ z, const float* __restrict__ WihP,
    const float* __restrict__ blP, float* __restrict__ G) {
  __shared__ float zt[16][260];  // pitch 260: 16B rows, conflict-free v2f frags
  const int tid = threadIdx.x;
  const int w = tid >> 5, lane = tid & 31, l16 = lane & 15, hi = lane >> 4;
  const int m0 = blockIdx.x * 16;
  const int nb = blockIdx.y * 256 + w * 64;
  const int dz = blockIdx.z;
  for (int idx = tid * 4; idx < 16 * HH; idx += 128 * 4) {
    const int r = idx >> 8, cc = idx & 255;
    *(float4*)&zt[r][cc] = *(const float4*)(z + (size_t)(m0 + r) * HH + cc);
  }
  __syncthreads();
  const float* W  = WihP + (size_t)dz * HH * G4H;
  const float* bl = blP + (size_t)dz * G4H;
  float* Gd = G + (size_t)dz * MM * G4H;
  v8f acc[4] = {};
  for (int k0 = 0; k0 < HH; k0 += 4) {
    const int ka = k0 + 2 * hi;
    v2f a = *(const v2f*)&zt[l16][ka];
#pragma unroll
    for (int s = 0; s < 4; ++s) {
      const int col = nb + 16 * s + l16;
      v2f b;
      b.x = W[(size_t)ka * G4H + col];
      b.y = W[(size_t)(ka + 1) * G4H + col];
      acc[s] = wmma_f32_k4(a, b, acc[s]);
    }
  }
#pragma unroll
  for (int s = 0; s < 4; ++s) {
    const int col = nb + 16 * s + l16;
    const float bias = bl[col];
#pragma unroll
    for (int vi = 0; vi < 8; ++vi) {
      const int row = m0 + vi + 8 * hi;
      Gd[(size_t)row * G4H + col] = acc[s][vi] + bias;
    }
  }
}

// ---------------- kernel 3: persistent LSTM scan (1 block per direction) ----
// 512 threads = 16 waves. Wave w owns gate columns [64w,64w+64) (permuted =>
// complete i/f/g/o for hidden units [16w,16w+16)). h kept in LDS as bf16;
// c/h/pool in registers. Each wave async-copies its 8KB slice of G[:,t,:] to
// LDS (GLOBAL_LOAD_ASYNC_TO_LDS_B128, ASYNCcnt) overlapping the WMMA phase.
#define HPITCH 264  // bf16 elements per h row (16B-aligned rows)
__global__ __launch_bounds__(512) void k_scan(
    const __bf16* __restrict__ WhhPT_all, const float* __restrict__ G_all,
    const int* __restrict__ mask, const int* __restrict__ lengths,
    float* __restrict__ fc_in) {
  __shared__ __bf16 h_lds[BB][HPITCH];      // ~16.9 KB
  __shared__ float g_lds[16 * BB * 64];     // 128 KB: [wave][b][64 gate cols]
  __shared__ int m_lds[BB];
  const int dir = blockIdx.x;
  const __bf16* Wt = WhhPT_all + (size_t)dir * G4H * HH;
  const float* G = G_all + (size_t)dir * MM * G4H;
  const int tid = threadIdx.x;
  const int w = tid >> 5, lane = tid & 31, l16 = lane & 15, hi = lane >> 4;
  const int j = 16 * w + l16;

  float c[16], h[16], pool[16], invlen[16];
#pragma unroll
  for (int tm = 0; tm < 2; ++tm)
#pragma unroll
    for (int vi = 0; vi < 8; ++vi) {
      const int q = tm * 8 + vi;
      const int b = 16 * tm + 8 * hi + vi;
      c[q] = h[q] = pool[q] = 0.f;
      invlen[q] = 1.0f / (float)lengths[b];
    }
  for (int i = tid; i < BB * HPITCH; i += 512) (&h_lds[0][0])[i] = (__bf16)0.f;
  __syncthreads();

  for (int step = 0; step < TT; ++step) {
    const int t = dir ? (TT - 1 - step) : step;
    // --- async stage: this wave's 32x64 G slice -> LDS (256B per lane) ---
    {
      const int b = lane;  // one batch row per lane
      const float* gsrc = G + ((size_t)b * TT + t) * G4H + 64 * w;
      const unsigned ldst = (unsigned)(unsigned long long)(const void*)
                                &g_lds[(w * BB + b) * 64];
      const unsigned long long gaddr = (unsigned long long)(const void*)gsrc;
      asm volatile(
          "global_load_async_to_lds_b128 %0, %1, off\n\t"
          "global_load_async_to_lds_b128 %0, %1, off offset:16\n\t"
          "global_load_async_to_lds_b128 %0, %1, off offset:32\n\t"
          "global_load_async_to_lds_b128 %0, %1, off offset:48\n\t"
          "global_load_async_to_lds_b128 %0, %1, off offset:64\n\t"
          "global_load_async_to_lds_b128 %0, %1, off offset:80\n\t"
          "global_load_async_to_lds_b128 %0, %1, off offset:96\n\t"
          "global_load_async_to_lds_b128 %0, %1, off offset:112\n\t"
          "global_load_async_to_lds_b128 %0, %1, off offset:128\n\t"
          "global_load_async_to_lds_b128 %0, %1, off offset:144\n\t"
          "global_load_async_to_lds_b128 %0, %1, off offset:160\n\t"
          "global_load_async_to_lds_b128 %0, %1, off offset:176\n\t"
          "global_load_async_to_lds_b128 %0, %1, off offset:192\n\t"
          "global_load_async_to_lds_b128 %0, %1, off offset:208\n\t"
          "global_load_async_to_lds_b128 %0, %1, off offset:224\n\t"
          "global_load_async_to_lds_b128 %0, %1, off offset:240"
          :: "v"(ldst), "v"(gaddr) : "memory");
    }
    if (tid < BB) m_lds[tid] = mask[tid * TT + t];

    // --- GEMM: raw_gates += h @ WhhPT (bf16 inputs, f32 accum) ---
    v8f acc[2][4] = {};
    for (int kb = 0; kb < HH; kb += 32) {
      v16bf a[2];
#pragma unroll
      for (int tm = 0; tm < 2; ++tm) {
        const __bf16* row = &h_lds[16 * tm + l16][0];
        const int kA = kb + 8 * hi;
#pragma unroll
        for (int e = 0; e < 8; ++e) a[tm][e] = row[kA + e];
#pragma unroll
        for (int e = 0; e < 8; ++e) a[tm][8 + e] = row[kA + 16 + e];
      }
#pragma unroll
      for (int s = 0; s < 4; ++s) {
        const __bf16* bp = Wt + (size_t)(64 * w + 16 * s + l16) * HH + kb + 8 * hi;
        v16bf bfrag;
#pragma unroll
        for (int e = 0; e < 16; ++e) bfrag[e] = bp[e + (e & 8)];
        acc[0][s] = wmma_bf16_k32(a[0], bfrag, acc[0][s]);
        acc[1][s] = wmma_bf16_k32(a[1], bfrag, acc[1][s]);
      }
    }
    asm volatile("s_wait_asynccnt 0x0" ::: "memory");  // this wave's G slice landed
    __syncthreads();  // all waves done reading h_lds; m_lds visible

    // --- pointwise LSTM cell, gates from registers + LDS-staged G ---
#pragma unroll
    for (int tm = 0; tm < 2; ++tm)
#pragma unroll
      for (int vi = 0; vi < 8; ++vi) {
        const int q = tm * 8 + vi;
        const int b = 16 * tm + 8 * hi + vi;
        const float* gw = &g_lds[(w * BB + b) * 64];
        const float ri = acc[tm][0][vi] + gw[l16];
        const float rf = acc[tm][1][vi] + gw[16 + l16];
        const float rg = acc[tm][2][vi] + gw[32 + l16];
        const float ro = acc[tm][3][vi] + gw[48 + l16];
        const float ig = sigmoidf_(ri);
        const float fg = sigmoidf_(rf);
        const float gg = tanhf(rg);
        const float og = sigmoidf_(ro);
        const float cn = fg * c[q] + ig * gg;
        const float hn = og * tanhf(cn);
        const float m = (float)m_lds[b];
        c[q] = m * cn + (1.f - m) * c[q];
        h[q] = m * hn + (1.f - m) * h[q];
        pool[q] += m * h[q] * invlen[q];
        h_lds[b][j] = (__bf16)h[q];
      }
    __syncthreads();  // h_lds updated before next step's GEMM
  }
#pragma unroll
  for (int tm = 0; tm < 2; ++tm)
#pragma unroll
    for (int vi = 0; vi < 8; ++vi) {
      const int q = tm * 8 + vi;
      const int b = 16 * tm + 8 * hi + vi;
      fc_in[(size_t)b * (2 * HH) + dir * HH + j] = pool[q];
    }
}

// ---------------- kernel 4: FC head ----------------
__global__ __launch_bounds__(256) void k_fc(
    const float* __restrict__ fc_in, const float* __restrict__ Wfc,
    const float* __restrict__ bfc, const float* __restrict__ Wout,
    const float* __restrict__ bout, float* __restrict__ out) {
  __shared__ float hid[BB][HH + 2];
  const int tid = threadIdx.x;
  const int w = tid >> 5, lane = tid & 31, l16 = lane & 15, hi = lane >> 4;
  v8f acc[2][2] = {};
  for (int k0 = 0; k0 < 2 * HH; k0 += 4) {
    const int ka = k0 + 2 * hi;
    v2f a0 = *(const v2f*)(fc_in + (size_t)l16 * (2 * HH) + ka);
    v2f a1 = *(const v2f*)(fc_in + (size_t)(16 + l16) * (2 * HH) + ka);
#pragma unroll
    for (int s = 0; s < 2; ++s) {
      const int col = (2 * w + s) * 16 + l16;
      v2f b;
      b.x = Wfc[(size_t)ka * HH + col];
      b.y = Wfc[(size_t)(ka + 1) * HH + col];
      acc[0][s] = wmma_f32_k4(a0, b, acc[0][s]);
      acc[1][s] = wmma_f32_k4(a1, b, acc[1][s]);
    }
  }
#pragma unroll
  for (int s = 0; s < 2; ++s) {
    const int col = (2 * w + s) * 16 + l16;
    const float bias = bfc[col];
#pragma unroll
    for (int tm = 0; tm < 2; ++tm)
#pragma unroll
      for (int vi = 0; vi < 8; ++vi) {
        const int row = 16 * tm + vi + 8 * hi;
        float v = acc[tm][s][vi] + bias;
        hid[row][col] = v > 0.f ? v : 0.f;
      }
  }
  __syncthreads();
  if (tid < BB * 2) {
    const int b = tid >> 1, cls = tid & 1;
    float sum = bout[cls];
    for (int k = 0; k < HH; ++k) sum += hid[b][k] * Wout[k * 2 + cls];
    out[b * 2 + cls] = sum;
  }
}

// ---------------- host launch ----------------
extern "C" void kernel_launch(void* const* d_in, const int* in_sizes, int n_in,
                              void* d_out, int out_size, void* d_ws, size_t ws_size,
                              hipStream_t stream) {
  (void)in_sizes; (void)n_in; (void)out_size; (void)ws_size;
  const float* emb   = (const float*)d_in[0];
  const float* Win   = (const float*)d_in[1];
  const float* bin   = (const float*)d_in[2];
  const float* Wih_f = (const float*)d_in[3];
  const float* Whh_f = (const float*)d_in[4];
  const float* bl_f  = (const float*)d_in[5];
  const float* Wih_b = (const float*)d_in[6];
  const float* Whh_b = (const float*)d_in[7];
  const float* bl_b  = (const float*)d_in[8];
  const float* Wfc   = (const float*)d_in[9];
  const float* bfc   = (const float*)d_in[10];
  const float* Wout  = (const float*)d_in[11];
  const float* bout  = (const float*)d_in[12];
  const int*   x     = (const int*)d_in[13];
  const int*   mask  = (const int*)d_in[14];
  const int*   lens  = (const int*)d_in[15];
  float* out = (float*)d_out;

  char* ws = (char*)d_ws;
  size_t off = 0;
  auto walloc = [&](size_t bytes) {
    void* p = ws + off;
    off = (off + bytes + 255) & ~(size_t)255;
    return p;
  };
  float*  z     = (float*)walloc((size_t)MM * HH * 4);          // 16 MB
  float*  G     = (float*)walloc((size_t)2 * MM * G4H * 4);     // 128 MB
  float*  WihP  = (float*)walloc((size_t)2 * HH * G4H * 4);     // 2 MB
  __bf16* WhhPT = (__bf16*)walloc((size_t)2 * G4H * HH * 2);    // 1 MB
  float*  blP   = (float*)walloc((size_t)2 * G4H * 4);
  float*  fcin  = (float*)walloc((size_t)BB * 2 * HH * 4);

  k_prep<<<2048, 256, 0, stream>>>(Wih_f, Whh_f, bl_f, Wih_b, Whh_b, bl_b,
                                   WihP, WhhPT, blP);
  k_embed_gemm<<<MM / 16, 128, 0, stream>>>(emb, Win, bin, x, z);
  k_gate_gemm<<<dim3(MM / 16, G4H / 256, 2), 128, 0, stream>>>(z, WihP, blP, G);
  k_scan<<<2, 512, 0, stream>>>(WhhPT, G, mask, lens, fcin);
  k_fc<<<1, 256, 0, stream>>>(fcin, Wfc, bfc, Wout, bout, out);
}
